// Layer_VanillaMHA_89747636617438
// MI455X (gfx1250) — matrix operand order repaired and checked by reading, not compiled, not run on hardware
//
#include <hip/hip_runtime.h>
#include <math.h>

// ---------------------------------------------------------------------------
// Types for CDNA5 WMMA (gfx1250, wave32)
// ---------------------------------------------------------------------------
typedef __attribute__((ext_vector_type(16))) __bf16 v16bf;
typedef __attribute__((ext_vector_type(8)))  float  v8f;
typedef __attribute__((ext_vector_type(8)))  unsigned int u32x8;

#define N_NODES   4096
#define IN_DIM    512
#define KDIM      64
#define OUT_DIM   512
#define N_HEADS   8

__device__ __forceinline__ unsigned short f2bf(float f) {
    unsigned u = __builtin_bit_cast(unsigned, f);
    u += 0x7FFFu + ((u >> 16) & 1u);          // round-to-nearest-even
    return (unsigned short)(u >> 16);
}

__device__ __forceinline__ unsigned pack2bf(float lo, float hi) {
    return (unsigned)f2bf(lo) | ((unsigned)f2bf(hi) << 16);
}

__device__ __forceinline__ v8f wmma_bf16(v16bf a, v16bf b, v8f c) {
    // D = A(16x32) * B(32x16) + C, f32 accumulate
    return __builtin_amdgcn_wmma_f32_16x16x32_bf16(
        /*neg_a=*/false, a, /*neg_b=*/false, b,
        /*c_mod=*/(short)0, c, /*reuse_a=*/false, /*reuse_b=*/false);
}

// A-matrix fragment, 16x32 bf16, row-major source with leading dim `ld`.
// Lane L holds row m=L&15; element i -> k = i + 16*(L>>4).
// All element pairs are k-contiguous -> coalesces to 2x b128 per lane.
__device__ __forceinline__ v16bf load_a16x32(const unsigned short* __restrict__ p,
                                             int ld, int lane) {
    const int m = lane & 15;
    const int h8 = (lane >> 4) * 8;
    const unsigned int* row = (const unsigned int*)(p + (size_t)m * ld);
    u32x8 u;
#pragma unroll
    for (int j = 0; j < 4; ++j) u[j]     = row[(2 * h8 + 2 * j) >> 1];
#pragma unroll
    for (int j = 0; j < 4; ++j) u[4 + j] = row[(2 * h8 + 8 + 2 * j) >> 1];
    return __builtin_bit_cast(v16bf, u);
}

// B-matrix fragment, 32x16 bf16, taken from a TRANSPOSED source BT[n][k]
// with leading dim `ld`. Lane L holds column n=L&15; element i -> k = i + 16*(L>>4).
__device__ __forceinline__ v16bf load_b32x16_fromT(const unsigned short* __restrict__ p,
                                                   int ld, int lane) {
    const int n = lane & 15;
    const int h16 = (lane >> 4) * 16;
    const unsigned int* row = (const unsigned int*)(p + (size_t)n * ld + h16);
    u32x8 u;
#pragma unroll
    for (int j = 0; j < 8; ++j) u[j] = row[j];
    return __builtin_bit_cast(v16bf, u);
}

// ---------------------------------------------------------------------------
// K0: conversions / transposes to bf16, and adjacency -> bitmask
// ---------------------------------------------------------------------------
__global__ void conv_f32_bf16(const float* __restrict__ src,
                              unsigned short* __restrict__ dst, int n) {
    int i = blockIdx.x * blockDim.x + threadIdx.x;
    if (i < n) dst[i] = f2bf(src[i]);
}

// Wk/Wq/Wv: [8][512][64] f32 -> [8][64][512] bf16 (n-major, k contiguous)
__global__ void conv_w_transpose(const float* __restrict__ src,
                                 unsigned short* __restrict__ dst) {
    int i = blockIdx.x * blockDim.x + threadIdx.x;   // n = 8*512*64
    int h = i >> 15;
    int rem = i & 32767;
    int k = rem >> 6;
    int n = rem & 63;
    dst[((size_t)h * 64 + n) * 512 + k] = f2bf(src[i]);
}

// Wo: [8][64][512] = [j=h*64+k][o] f32 -> WoT[o][j] bf16
__global__ void conv_wo_transpose(const float* __restrict__ src,
                                  unsigned short* __restrict__ dst) {
    int i = blockIdx.x * blockDim.x + threadIdx.x;   // n = 512*512
    int j = i >> 9;
    int o = i & 511;
    dst[(size_t)o * 512 + j] = f2bf(src[i]);
}

// adjacency [4096][4096] f32 -> bitmask [4096][128] u32 (self-loops forced).
// One wave per 32-bit word; fully coalesced reads, ballot to pack.
__global__ void pack_mask(const float* __restrict__ adj,
                          unsigned* __restrict__ amask) {
    int gid = blockIdx.x * blockDim.x + threadIdx.x;
    int wid = gid >> 5;           // word id = row*128 + word
    int lane = gid & 31;
    int row = wid >> 7;
    int col = ((wid & 127) << 5) + lane;
    float a = adj[(size_t)row * N_NODES + col];
    bool pred = (a != 0.0f) || (row == col);
    unsigned long long b = __ballot(pred);
    if (lane == 0) amask[wid] = (unsigned)b;
}

// ---------------------------------------------------------------------------
// K1: projections  P[h] = X(4096x512) @ W[h](512x64)
//   proj 0 -> Kb[h][n][64], proj 1 -> Qb (scaled by 1/8), proj 2 -> VbT[h][64][n]
// ---------------------------------------------------------------------------
__global__ __launch_bounds__(256) void proj_kernel(
    const unsigned short* __restrict__ Xb,     // [4096][512]
    const unsigned short* __restrict__ WT,     // [24][64][512]
    unsigned short* __restrict__ Qb,           // [8][4096][64]
    unsigned short* __restrict__ Kb,           // [8][4096][64]
    unsigned short* __restrict__ VbT)          // [8][64][4096]
{
    const int lane = threadIdx.x & 31;
    const int wave = threadIdx.x >> 5;
    const int y = blockIdx.y;
    const int proj = y >> 3, head = y & 7;
    const int rbase = blockIdx.x * 128 + wave * 16;
    const unsigned short* W = WT + (size_t)y * 64 * 512;

    v8f c[4] = {v8f{}, v8f{}, v8f{}, v8f{}};
    for (int k0 = 0; k0 < 512; k0 += 32) {
        v16bf a = load_a16x32(Xb + (size_t)rbase * 512 + k0, 512, lane);
#pragma unroll
        for (int nt = 0; nt < 4; ++nt) {
            v16bf b = load_b32x16_fromT(W + (size_t)(nt * 16) * 512 + k0, 512, lane);
            c[nt] = wmma_bf16(a, b, c[nt]);
        }
    }
    const float scale = (proj == 1) ? 0.125f : 1.0f;   // 1/sqrt(64) folded into Q
    const int n0 = lane & 15, half = lane >> 4;
#pragma unroll
    for (int nt = 0; nt < 4; ++nt)
#pragma unroll
        for (int r = 0; r < 8; ++r) {
            int row = rbase + r + 8 * half;
            int col = nt * 16 + n0;
            unsigned short v = f2bf(c[nt][r] * scale);
            if (proj == 0)      Kb[((size_t)head * N_NODES + row) * 64 + col] = v;
            else if (proj == 1) Qb[((size_t)head * N_NODES + row) * 64 + col] = v;
            else                VbT[((size_t)head * 64 + col) * N_NODES + row] = v;
        }
}

// ---------------------------------------------------------------------------
// K2: fused masked flash attention.
// Block = 256 thr = 8 waves; wave w = head w, handling 32 queries (2 tiles of
// 16) so every K/V fragment load feeds two S^T / PV WMMAs (doubles K/V reuse).
// S^T = K·Q^T puts each query in a fixed lane -> per-query softmax needs only
// in-register reduces + one shfl_xor(16). P^T fragments for O^T = V^T·P^T are
// assembled with bf16-pair cross-lane shuffles (uniform register indices).
// Adjacency handled as a packed bitmask: one dword per 32 keys per query.
// ---------------------------------------------------------------------------
__global__ __launch_bounds__(256) void attn_kernel(
    const unsigned short* __restrict__ Qb,
    const unsigned short* __restrict__ Kb,
    const unsigned short* __restrict__ VbT,
    const unsigned* __restrict__ amask,        // [4096][128]
    unsigned short* __restrict__ Ob)           // [4096][512] (cols = head*64+v)
{
    const int lane = threadIdx.x & 31;
    const int head = threadIdx.x >> 5;
    const int qbase = blockIdx.x * 32;
    const int qcol = lane & 15;
    const int half = lane >> 4;

    const unsigned short* Qh = Qb + (size_t)head * N_NODES * 64;
    const unsigned short* Kh = Kb + (size_t)head * N_NODES * 64;
    const unsigned short* Vh = VbT + (size_t)head * 64 * N_NODES;

    const unsigned* am0 = amask + (size_t)(qbase + qcol) * 128;
    const unsigned* am1 = amask + (size_t)(qbase + 16 + qcol) * 128;

    // Q^T B-fragments (fixed for whole key loop): B[d][q] = Q[q][d]
    v16bf bq[2][2];
#pragma unroll
    for (int t = 0; t < 2; ++t) {
        bq[t][0] = load_b32x16_fromT(Qh + (size_t)(qbase + 16 * t) * 64 + 0, 64, lane);
        bq[t][1] = load_b32x16_fromT(Qh + (size_t)(qbase + 16 * t) * 64 + 32, 64, lane);
    }

    v8f o[2][4];
#pragma unroll
    for (int t = 0; t < 2; ++t)
#pragma unroll
        for (int vt = 0; vt < 4; ++vt) o[t][vt] = v8f{};
    float m_run[2] = {-1.0e30f, -1.0e30f};
    float l_run[2] = {0.0f, 0.0f};

    for (int kb = 0; kb < N_NODES; kb += 32) {
        if (kb + 32 < N_NODES) {
            __builtin_prefetch(Kh + (size_t)(kb + 32) * 64, 0, 0);
            __builtin_prefetch(Vh + kb + 32, 0, 0);
        }
        float s0[2][8], s1[2][8];   // raw scores, subtile 0 / 1
        {
            v16bf ak0 = load_a16x32(Kh + (size_t)kb * 64 + 0, 64, lane);
            v16bf ak1 = load_a16x32(Kh + (size_t)kb * 64 + 32, 64, lane);
#pragma unroll
            for (int t = 0; t < 2; ++t) {
                v8f s = v8f{};
                s = wmma_bf16(ak0, bq[t][0], s);
                s = wmma_bf16(ak1, bq[t][1], s);
#pragma unroll
                for (int r = 0; r < 8; ++r) s0[t][r] = s[r];
            }
        }
        {
            v16bf ak0 = load_a16x32(Kh + (size_t)(kb + 16) * 64 + 0, 64, lane);
            v16bf ak1 = load_a16x32(Kh + (size_t)(kb + 16) * 64 + 32, 64, lane);
#pragma unroll
            for (int t = 0; t < 2; ++t) {
                v8f s = v8f{};
                s = wmma_bf16(ak0, bq[t][0], s);
                s = wmma_bf16(ak1, bq[t][1], s);
#pragma unroll
                for (int r = 0; r < 8; ++r) s1[t][r] = s[r];
            }
        }

        // ---- mask + online softmax + bf16-pair packing, per q-tile ----
        unsigned pk0[2][4], pk1[2][4];
#pragma unroll
        for (int t = 0; t < 2; ++t) {
            const unsigned mwr = (t ? am1 : am0)[kb >> 5];
            const unsigned mw = mwr >> (8 * half);   // p0 bits at r, p1 bits at 16+r
            const float NEG = -__builtin_inff();
#pragma unroll
            for (int r = 0; r < 8; ++r) {
                if (!((mw >> r) & 1u))        s0[t][r] = NEG;
                if (!((mw >> (16 + r)) & 1u)) s1[t][r] = NEG;
            }
            float bm = fmaxf(s0[t][0], s1[t][0]);
#pragma unroll
            for (int r = 1; r < 8; ++r) bm = fmaxf(bm, fmaxf(s0[t][r], s1[t][r]));
            bm = fmaxf(bm, __shfl_xor(bm, 16, 32));
            float m_new = fmaxf(m_run[t], bm);
            float alpha = __expf(m_run[t] - m_new);
            float lsum = 0.0f;
#pragma unroll
            for (int r = 0; r < 8; ++r) {
                s0[t][r] = __expf(s0[t][r] - m_new); lsum += s0[t][r];
                s1[t][r] = __expf(s1[t][r] - m_new); lsum += s1[t][r];
            }
            lsum += __shfl_xor(lsum, 16, 32);
            l_run[t] = l_run[t] * alpha + lsum;
            m_run[t] = m_new;
#pragma unroll
            for (int vt = 0; vt < 4; ++vt)
#pragma unroll
                for (int r = 0; r < 8; ++r) o[t][vt][r] *= alpha;
#pragma unroll
            for (int j = 0; j < 4; ++j) {
                pk0[t][j] = pack2bf(s0[t][2 * j], s0[t][2 * j + 1]);
                pk1[t][j] = pack2bf(s1[t][2 * j], s1[t][2 * j + 1]);
            }
        }

        // ---- assemble P^T B-fragments: B[m][q] = P[q][kb+m] ----
        // element pair j: src lane = qcol + (j>=4 ? 16 : 0); sub picked by our
        // own half (upper-half lanes need subtile-1 probabilities).
        v16bf bp[2];
#pragma unroll
        for (int t = 0; t < 2; ++t) {
            u32x8 bu;
#pragma unroll
            for (int j = 0; j < 8; ++j) {
                int srcl = qcol + ((j >= 4) ? 16 : 0);
                unsigned a = __shfl(pk0[t][j & 3], srcl, 32);
                unsigned b = __shfl(pk1[t][j & 3], srcl, 32);
                bu[j] = half ? b : a;
            }
            bp[t] = __builtin_bit_cast(v16bf, bu);
        }

        // ---- O^T += V^T(64x32) · P^T(32x16), V fragments shared by tiles ----
#pragma unroll
        for (int vt = 0; vt < 4; ++vt) {
            v16bf av = load_a16x32(Vh + (size_t)(vt * 16) * N_NODES + kb, N_NODES, lane);
#pragma unroll
            for (int t = 0; t < 2; ++t) o[t][vt] = wmma_bf16(av, bp[t], o[t][vt]);
        }
    }

#pragma unroll
    for (int t = 0; t < 2; ++t) {
        const float inv = 1.0f / l_run[t];   // > 0 via forced self-loop
        unsigned* obase =
            (unsigned*)(Ob + (size_t)(qbase + 16 * t + qcol) * 512 + head * 64);
#pragma unroll
        for (int vt = 0; vt < 4; ++vt)
#pragma unroll
            for (int j = 0; j < 4; ++j) {
                unsigned u = pack2bf(o[t][vt][2 * j] * inv, o[t][vt][2 * j + 1] * inv);
                obase[(vt * 16 + 8 * half + 2 * j) >> 1] = u;
            }
    }
}

// ---------------------------------------------------------------------------
// K3: out = Ob(4096x512) @ Wo_concat(512x512), f32 output. Grid (32, 8).
// ---------------------------------------------------------------------------
__global__ __launch_bounds__(256) void outproj_kernel(
    const unsigned short* __restrict__ Ob,     // [4096][512]
    const unsigned short* __restrict__ WoT,    // [512][512] o-major
    float* __restrict__ out)                   // [4096][512]
{
    const int lane = threadIdx.x & 31;
    const int wave = threadIdx.x >> 5;
    const int rbase = blockIdx.x * 128 + wave * 16;
    const int ob = blockIdx.y * 64;

    v8f c[4] = {v8f{}, v8f{}, v8f{}, v8f{}};
    for (int k0 = 0; k0 < 512; k0 += 32) {
        v16bf a = load_a16x32(Ob + (size_t)rbase * 512 + k0, 512, lane);
#pragma unroll
        for (int nt = 0; nt < 4; ++nt) {
            v16bf b = load_b32x16_fromT(WoT + (size_t)(ob + nt * 16) * 512 + k0, 512, lane);
            c[nt] = wmma_bf16(a, b, c[nt]);
        }
    }
    const int n0 = lane & 15, half = lane >> 4;
#pragma unroll
    for (int nt = 0; nt < 4; ++nt)
#pragma unroll
        for (int r = 0; r < 8; ++r)
            out[(size_t)(rbase + r + 8 * half) * 512 + ob + nt * 16 + n0] = c[nt][r];
}

// ---------------------------------------------------------------------------
// Host launcher
// ---------------------------------------------------------------------------
extern "C" void kernel_launch(void* const* d_in, const int* in_sizes, int n_in,
                              void* d_out, int out_size, void* d_ws, size_t ws_size,
                              hipStream_t stream) {
    const float* node = (const float*)d_in[0];   // [4096][512]
    const float* adj  = (const float*)d_in[1];   // [4096][4096]
    const float* Wks  = (const float*)d_in[2];   // [8][512][64]
    const float* Wqs  = (const float*)d_in[3];
    const float* Wvs  = (const float*)d_in[4];
    const float* Wo   = (const float*)d_in[5];   // [8][64][512]
    float* out = (float*)d_out;

    unsigned short* w = (unsigned short*)d_ws;
    const size_t XN  = (size_t)N_NODES * IN_DIM;          // 2,097,152
    const size_t WN  = (size_t)N_HEADS * IN_DIM * KDIM;   //   262,144 per proj
    const size_t QKN = (size_t)N_HEADS * N_NODES * KDIM;  // 2,097,152
    unsigned short* Xb  = w;                 w += XN;
    unsigned short* WT  = w;                 w += 3 * WN;
    unsigned short* WoT = w;                 w += (size_t)512 * 512;
    unsigned short* Qb  = w;                 w += QKN;
    unsigned short* Kb  = w;                 w += QKN;
    unsigned short* VbT = w;                 w += QKN;
    unsigned short* Ob  = w;                 w += XN;      // [4096][512]
    unsigned* amask = (unsigned*)w;                        // [4096][128] u32

    // K0: pack / transpose to bf16; adjacency -> bitmask with self-loops
    conv_f32_bf16<<<(int)((XN + 255) / 256), 256, 0, stream>>>(node, Xb, (int)XN);
    conv_w_transpose<<<(int)(WN / 256), 256, 0, stream>>>(Wks, WT + 0 * WN);
    conv_w_transpose<<<(int)(WN / 256), 256, 0, stream>>>(Wqs, WT + 1 * WN);
    conv_w_transpose<<<(int)(WN / 256), 256, 0, stream>>>(Wvs, WT + 2 * WN);
    conv_wo_transpose<<<(512 * 512) / 256, 256, 0, stream>>>(Wo, WoT);
    pack_mask<<<(N_NODES * 128 * 32) / 256, 256, 0, stream>>>(adj, amask);

    // K1: Q/K/V projections (24 head-proj GEMMs)
    proj_kernel<<<dim3(N_NODES / 128, 24), 256, 0, stream>>>(Xb, WT, Qb, Kb, VbT);

    // K2: fused masked flash attention (32 queries x 8 heads per CTA)
    attn_kernel<<<N_NODES / 32, 256, 0, stream>>>(Qb, Kb, VbT, amask, Ob);

    // K3: output projection summed over heads
    outproj_kernel<<<dim3(N_NODES / 128, 8), 256, 0, stream>>>(Ob, WoT, out);
}
